// CamProxyLoss_15710990369514
// MI455X (gfx1250) — matrix operand order repaired and checked by reading, not compiled, not run on hardware
//
#include <hip/hip_runtime.h>
#include <stdint.h>

// CDNA5 wave32 WMMA types
typedef __attribute__((ext_vector_type(16))) __bf16 v16bf;
typedef __attribute__((ext_vector_type(8)))  float  v8f;

#define NUM_CAMS 15
#define BK   32
#define LDSS 40   // ushort row stride in LDS (32 + 8 pad) -> conflict-spread b128 reads

struct Q2 { uint4 a, b; };

__device__ __forceinline__ unsigned bf16rn(float x) {
  unsigned u = __float_as_uint(x);
  return (u + 0x7FFFu + ((u >> 16) & 1u)) >> 16;   // round-to-nearest-even bf16
}

__device__ __forceinline__ v16bf ld_frag(const unsigned short* plane, int row, int c0, int c1) {
  Q2 q;
  q.a = *(const uint4*)(plane + row * LDSS + c0);
  q.b = *(const uint4*)(plane + row * LDSS + c1);
  return __builtin_bit_cast(v16bf, q);
}

// ---------------------------------------------------------------------------
// Preprocess: split fp32 matrix into bf16 hi/lo planes (x = hi + lo, bf16x3)
// ---------------------------------------------------------------------------
__global__ void cam_proxy_split(const float* __restrict__ src,
                                unsigned short* __restrict__ hi,
                                unsigned short* __restrict__ lo, long n4)
{
  long i = (long)blockIdx.x * blockDim.x + threadIdx.x;
  if (i >= n4) return;
  float4 f = ((const float4*)src)[i];
  unsigned h0 = bf16rn(f.x), h1 = bf16rn(f.y), h2 = bf16rn(f.z), h3 = bf16rn(f.w);
  float l0 = f.x - __uint_as_float(h0 << 16);
  float l1 = f.y - __uint_as_float(h1 << 16);
  float l2 = f.z - __uint_as_float(h2 << 16);
  float l3 = f.w - __uint_as_float(h3 << 16);
  ((uint2*)hi)[i] = make_uint2(h0 | (h1 << 16), h2 | (h3 << 16));
  ((uint2*)lo)[i] = make_uint2(bf16rn(l0) | (bf16rn(l1) << 16),
                               bf16rn(l2) | (bf16rn(l3) << 16));
}

// ---------------------------------------------------------------------------
// Main path: 128x256 block tile, 8 waves each owning 64x64 (16 WMMA tiles).
// bf16x3 split GEMM fused with shifted exp row-sum (logsumexp numerator).
// Double-buffered LDS: one barrier per k-step, staging overlaps the WMMA burst.
// ---------------------------------------------------------------------------
__global__ __launch_bounds__(256) void cam_proxy_gemm_pre(
    const unsigned short* __restrict__ fhi, const unsigned short* __restrict__ flo,
    const unsigned short* __restrict__ phi, const unsigned short* __restrict__ plo,
    const float* __restrict__ temp, float* __restrict__ sumexp,
    int B, int N, int D)
{
  __shared__ __align__(16) unsigned short sAhi[2][128 * LDSS];
  __shared__ __align__(16) unsigned short sAlo[2][128 * LDSS];
  __shared__ __align__(16) unsigned short sBhi[2][256 * LDSS];
  __shared__ __align__(16) unsigned short sBlo[2][256 * LDSS];

  const int tid  = threadIdx.x;
  const int lane = tid & 31;
  const int wave = tid >> 5;
  const int wr   = wave & 1;        // 2 row groups x 64 rows
  const int wc   = wave >> 1;       // 4 col groups x 64 cols
  const int m    = lane & 15;
  const int h    = (lane >> 4) & 1;

  const int rowBase = blockIdx.y * 128;
  const int colBase = blockIdx.x * 256;

  // loaders: A -> thread pair per row (32B each half), B -> one thread per row (64B)
  int arow = rowBase + (tid >> 1); if (arow >= B) arow = B - 1;
  int brow = colBase + tid;        if (brow >= N) brow = N - 1;
  const int ac = (tid & 1) * 16;                       // ushort offset within k-slice
  const unsigned short* gAh = fhi + (size_t)arow * D + ac;
  const unsigned short* gAl = flo + (size_t)arow * D + ac;
  const unsigned short* gBh = phi + (size_t)brow * D;
  const unsigned short* gBl = plo + (size_t)brow * D;
  const int sbA = (tid >> 1) * LDSS + ac;
  const int sbB = tid * LDSS;

  v8f acc[4][4] = {};
  uint4 st[12];

  // prologue: stage k-slice 0 into buffer 0
  st[0]  = *(const uint4*)(gAh);       st[1]  = *(const uint4*)(gAh + 8);
  st[2]  = *(const uint4*)(gAl);       st[3]  = *(const uint4*)(gAl + 8);
  st[4]  = *(const uint4*)(gBh);       st[5]  = *(const uint4*)(gBh + 8);
  st[6]  = *(const uint4*)(gBh + 16);  st[7]  = *(const uint4*)(gBh + 24);
  st[8]  = *(const uint4*)(gBl);       st[9]  = *(const uint4*)(gBl + 8);
  st[10] = *(const uint4*)(gBl + 16);  st[11] = *(const uint4*)(gBl + 24);
  *(uint4*)&sAhi[0][sbA]      = st[0];  *(uint4*)&sAhi[0][sbA + 8]  = st[1];
  *(uint4*)&sAlo[0][sbA]      = st[2];  *(uint4*)&sAlo[0][sbA + 8]  = st[3];
  *(uint4*)&sBhi[0][sbB]      = st[4];  *(uint4*)&sBhi[0][sbB + 8]  = st[5];
  *(uint4*)&sBhi[0][sbB + 16] = st[6];  *(uint4*)&sBhi[0][sbB + 24] = st[7];
  *(uint4*)&sBlo[0][sbB]      = st[8];  *(uint4*)&sBlo[0][sbB + 8]  = st[9];
  *(uint4*)&sBlo[0][sbB + 16] = st[10]; *(uint4*)&sBlo[0][sbB + 24] = st[11];
  __syncthreads();

  const int nk = D / BK;
  for (int kt = 0; kt < nk; ++kt) {
    const int cur  = kt & 1;
    const int nxt  = cur ^ 1;
    const bool more = (kt + 1) < nk;

    // issue next k-slice global loads first: a full WMMA burst covers the latency
    if (more) {
      const int ko = (kt + 1) * BK;
      st[0]  = *(const uint4*)(gAh + ko);       st[1]  = *(const uint4*)(gAh + ko + 8);
      st[2]  = *(const uint4*)(gAl + ko);       st[3]  = *(const uint4*)(gAl + ko + 8);
      st[4]  = *(const uint4*)(gBh + ko);       st[5]  = *(const uint4*)(gBh + ko + 8);
      st[6]  = *(const uint4*)(gBh + ko + 16);  st[7]  = *(const uint4*)(gBh + ko + 24);
      st[8]  = *(const uint4*)(gBl + ko);       st[9]  = *(const uint4*)(gBl + ko + 8);
      st[10] = *(const uint4*)(gBl + ko + 16);  st[11] = *(const uint4*)(gBl + ko + 24);
    }

    // A frag (16-bit 16x32): half 0 -> K {0..7,16..23}, half 1 -> K {8..15,24..31}
    v16bf ahi[4], alo[4];
    #pragma unroll
    for (int rt = 0; rt < 4; ++rt) {
      int r = wr * 64 + rt * 16 + m;
      ahi[rt] = ld_frag(sAhi[cur], r, h * 8, h * 8 + 16);
      alo[rt] = ld_frag(sAlo[cur], r, h * 8, h * 8 + 16);
    }
    // B frag (32x16): lane = column, half 0 -> K {0..7,16..23}, half 1 -> K {8..15,24..31}.
    // Loaded per-ct: one hi/lo pair feeds 12 WMMAs, keeping register pressure low.
    #pragma unroll
    for (int ct = 0; ct < 4; ++ct) {
      int c = wc * 64 + ct * 16 + m;
      v16bf bhi = ld_frag(sBhi[cur], c, h * 8, h * 8 + 16);
      v16bf blo = ld_frag(sBlo[cur], c, h * 8, h * 8 + 16);
      #pragma unroll
      for (int rt = 0; rt < 4; ++rt) {
        acc[rt][ct] = __builtin_amdgcn_wmma_f32_16x16x32_bf16(
            false, ahi[rt], false, bhi, (short)0, acc[rt][ct], false, false);
        acc[rt][ct] = __builtin_amdgcn_wmma_f32_16x16x32_bf16(
            false, alo[rt], false, bhi, (short)0, acc[rt][ct], false, false);
        acc[rt][ct] = __builtin_amdgcn_wmma_f32_16x16x32_bf16(
            false, ahi[rt], false, blo, (short)0, acc[rt][ct], false, false);
      }
    }

    // stage next slice into the other buffer (safe: last read before the previous barrier)
    if (more) {
      *(uint4*)&sAhi[nxt][sbA]      = st[0];  *(uint4*)&sAhi[nxt][sbA + 8]  = st[1];
      *(uint4*)&sAlo[nxt][sbA]      = st[2];  *(uint4*)&sAlo[nxt][sbA + 8]  = st[3];
      *(uint4*)&sBhi[nxt][sbB]      = st[4];  *(uint4*)&sBhi[nxt][sbB + 8]  = st[5];
      *(uint4*)&sBhi[nxt][sbB + 16] = st[6];  *(uint4*)&sBhi[nxt][sbB + 24] = st[7];
      *(uint4*)&sBlo[nxt][sbB]      = st[8];  *(uint4*)&sBlo[nxt][sbB + 8]  = st[9];
      *(uint4*)&sBlo[nxt][sbB + 16] = st[10]; *(uint4*)&sBlo[nxt][sbB + 24] = st[11];
    }
    __syncthreads();   // single barrier per k-step
  }

  // Epilogue: exp(logit - shift) row sums; shift = 1/temp bounds all logits.
  const float invT = 1.0f / temp[0];
  #pragma unroll
  for (int rt = 0; rt < 4; ++rt) {
    float rsum[8];
    #pragma unroll
    for (int v = 0; v < 8; ++v) rsum[v] = 0.0f;
    #pragma unroll
    for (int ct = 0; ct < 4; ++ct) {
      int  gc    = colBase + wc * 64 + ct * 16 + m;
      bool valid = gc < N;
      #pragma unroll
      for (int v = 0; v < 8; ++v)
        rsum[v] += valid ? __expf(acc[rt][ct][v] * invT - invT) : 0.0f;
    }
    #pragma unroll
    for (int v = 0; v < 8; ++v) {
      #pragma unroll
      for (int s = 1; s < 16; s <<= 1) rsum[v] += __shfl_xor(rsum[v], s, 32);
    }
    if (m == 0) {
      #pragma unroll
      for (int v = 0; v < 8; ++v) {
        int gr = rowBase + wr * 64 + rt * 16 + v + 8 * h;
        if (gr < B) atomicAdd(&sumexp[gr], rsum[v]);
      }
    }
  }
}

// ---------------------------------------------------------------------------
// Fallback (small workspace): fused convert-in-LDS variant, 128x128 tile.
// ---------------------------------------------------------------------------
__device__ __forceinline__ void split_store4(float4 f, unsigned short* hiP, unsigned short* loP) {
  unsigned h0 = bf16rn(f.x), h1 = bf16rn(f.y), h2 = bf16rn(f.z), h3 = bf16rn(f.w);
  float l0 = f.x - __uint_as_float(h0 << 16);
  float l1 = f.y - __uint_as_float(h1 << 16);
  float l2 = f.z - __uint_as_float(h2 << 16);
  float l3 = f.w - __uint_as_float(h3 << 16);
  *(uint2*)hiP = make_uint2(h0 | (h1 << 16), h2 | (h3 << 16));
  *(uint2*)loP = make_uint2(bf16rn(l0) | (bf16rn(l1) << 16), bf16rn(l2) | (bf16rn(l3) << 16));
}

__global__ __launch_bounds__(256) void cam_proxy_gemm_fused(
    const float* __restrict__ feats, const float* __restrict__ proxies,
    const float* __restrict__ temp, float* __restrict__ sumexp,
    int B, int N, int D)
{
  __shared__ __align__(16) unsigned short sAhi[128 * LDSS];
  __shared__ __align__(16) unsigned short sAlo[128 * LDSS];
  __shared__ __align__(16) unsigned short sBhi[128 * LDSS];
  __shared__ __align__(16) unsigned short sBlo[128 * LDSS];

  const int tid  = threadIdx.x;
  const int lane = tid & 31;
  const int wave = tid >> 5;
  const int wr   = wave & 3;
  const int wc   = wave >> 2;
  const int m    = lane & 15;
  const int h    = (lane >> 4) & 1;
  const int rowBase = blockIdx.y * 128;
  const int colBase = blockIdx.x * 128;
  const int lr = tid >> 1;
  const int lc = (tid & 1) * 16;

  int arow = rowBase + lr; if (arow >= B) arow = B - 1;
  int brow = colBase + lr; if (brow >= N) brow = N - 1;
  const float4* Ag = (const float4*)(feats   + (size_t)arow * D + lc);
  const float4* Bg = (const float4*)(proxies + (size_t)brow * D + lc);

  v8f acc[2][4] = {};
  float4 ra[4], rb[4], na[4], nb[4];
  #pragma unroll
  for (int i = 0; i < 4; ++i) { ra[i] = Ag[i]; rb[i] = Bg[i]; }

  const int sb = lr * LDSS + lc;
  const int nk = D / BK;
  for (int kt = 0; kt < nk; ++kt) {
    #pragma unroll
    for (int i = 0; i < 4; ++i) {
      split_store4(ra[i], &sAhi[sb + 4 * i], &sAlo[sb + 4 * i]);
      split_store4(rb[i], &sBhi[sb + 4 * i], &sBlo[sb + 4 * i]);
    }
    if (kt + 1 < nk) {
      int j = (kt + 1) * (BK / 4);
      #pragma unroll
      for (int i = 0; i < 4; ++i) { na[i] = Ag[j + i]; nb[i] = Bg[j + i]; }
    }
    __syncthreads();
    v16bf ahi[2], alo[2];
    #pragma unroll
    for (int rt = 0; rt < 2; ++rt) {
      int r = wr * 32 + rt * 16 + m;
      ahi[rt] = ld_frag(sAhi, r, h * 8, h * 8 + 16);
      alo[rt] = ld_frag(sAlo, r, h * 8, h * 8 + 16);
    }
    #pragma unroll
    for (int ct = 0; ct < 4; ++ct) {
      int c = wc * 64 + ct * 16 + m;
      v16bf bhi = ld_frag(sBhi, c, h * 8, h * 8 + 16);
      v16bf blo = ld_frag(sBlo, c, h * 8, h * 8 + 16);
      #pragma unroll
      for (int rt = 0; rt < 2; ++rt) {
        acc[rt][ct] = __builtin_amdgcn_wmma_f32_16x16x32_bf16(
            false, ahi[rt], false, bhi, (short)0, acc[rt][ct], false, false);
        acc[rt][ct] = __builtin_amdgcn_wmma_f32_16x16x32_bf16(
            false, alo[rt], false, bhi, (short)0, acc[rt][ct], false, false);
        acc[rt][ct] = __builtin_amdgcn_wmma_f32_16x16x32_bf16(
            false, ahi[rt], false, blo, (short)0, acc[rt][ct], false, false);
      }
    }
    __syncthreads();
    #pragma unroll
    for (int i = 0; i < 4; ++i) { ra[i] = na[i]; rb[i] = nb[i]; }
  }

  const float invT = 1.0f / temp[0];
  #pragma unroll
  for (int rt = 0; rt < 2; ++rt) {
    float rsum[8];
    #pragma unroll
    for (int v = 0; v < 8; ++v) rsum[v] = 0.0f;
    #pragma unroll
    for (int ct = 0; ct < 4; ++ct) {
      int  gc    = colBase + wc * 64 + ct * 16 + m;
      bool valid = gc < N;
      #pragma unroll
      for (int v = 0; v < 8; ++v)
        rsum[v] += valid ? __expf(acc[rt][ct][v] * invT - invT) : 0.0f;
    }
    #pragma unroll
    for (int v = 0; v < 8; ++v) {
      #pragma unroll
      for (int s = 1; s < 16; s <<= 1) rsum[v] += __shfl_xor(rsum[v], s, 32);
    }
    if (m == 0) {
      #pragma unroll
      for (int v = 0; v < 8; ++v) {
        int gr = rowBase + wr * 32 + rt * 16 + v + 8 * h;
        if (gr < B) atomicAdd(&sumexp[gr], rsum[v]);
      }
    }
  }
}

// Exact fp32 own logit: own[b] = dot(feats[b], proxies[labels[b]]) / temp
__global__ __launch_bounds__(256) void cam_proxy_own(
    const float* __restrict__ feats, const float* __restrict__ proxies,
    const int* __restrict__ labels, const float* __restrict__ temp,
    float* __restrict__ own, int B, int D)
{
  int wave = threadIdx.x >> 5;
  int lane = threadIdx.x & 31;
  int b = blockIdx.x * 8 + wave;
  if (b >= B) return;                      // wave-uniform
  int lab = labels[b];
  const float4* f = (const float4*)(feats   + (size_t)b   * D);
  const float4* p = (const float4*)(proxies + (size_t)lab * D);
  float acc = 0.0f;
  for (int j = lane; j < D / 4; j += 32) {
    float4 a = f[j], q = p[j];
    acc += a.x * q.x + a.y * q.y + a.z * q.z + a.w * q.w;
  }
  #pragma unroll
  for (int s = 1; s < 32; s <<= 1) acc += __shfl_xor(acc, s, 32);
  if (lane == 0) own[b] = acc / temp[0];
}

__global__ void cam_proxy_init(unsigned long long* minpack, float* sumexp,
                               float* lossAcc, int nseg, int B)
{
  int i = blockIdx.x * blockDim.x + threadIdx.x;
  if (i < nseg) minpack[i] = ~0ull;
  if (i < B)    sumexp[i]  = 0.0f;
  if (i < 2)    lossAcc[i] = 0.0f;
}

// argmin-with-first-index via single packed u64 atomicMin
__global__ void cam_proxy_argmin(const float* __restrict__ own,
                                 const int* __restrict__ labels,
                                 const int* __restrict__ cams,
                                 unsigned long long* minpack, int B)
{
  int b = blockIdx.x * blockDim.x + threadIdx.x;
  if (b >= B) return;
  unsigned u = __float_as_uint(own[b]);
  u = (u & 0x80000000u) ? ~u : (u | 0x80000000u);   // order-preserving float->uint
  unsigned long long key = ((unsigned long long)u << 32) | (unsigned)b;
  int g = labels[b] * NUM_CAMS + cams[b];
  atomicMin(&minpack[g], key);
}

__global__ void cam_proxy_pid(const unsigned long long* __restrict__ minpack,
                              const float* __restrict__ sumexp,
                              const float* __restrict__ own,
                              const float* __restrict__ temp,
                              float* lossAcc, int Np)
{
  int p = blockIdx.x * blockDim.x + threadIdx.x;
  if (p >= Np) return;
  float invT = 1.0f / temp[0];
  int cnt = 0; float acc = 0.0f;
  for (int c = 0; c < NUM_CAMS; ++c) {
    unsigned long long k = minpack[p * NUM_CAMS + c];
    if (k != ~0ull) {
      int idx = (int)(unsigned)(k & 0xFFFFFFFFu);
      acc += logf(sumexp[idx]) + invT - own[idx];   // logsumexp - own
      cnt++;
    }
  }
  if (cnt > 0) {
    atomicAdd(&lossAcc[0], acc / (float)cnt);
    atomicAdd(&lossAcc[1], 1.0f);
  }
}

__global__ void cam_proxy_final(const float* lossAcc, float* out)
{
  out[0] = lossAcc[0] / lossAcc[1];
}

extern "C" void kernel_launch(void* const* d_in, const int* in_sizes, int n_in,
                              void* d_out, int out_size, void* d_ws, size_t ws_size,
                              hipStream_t stream)
{
  const float* feats   = (const float*)d_in[0];
  const int*   labels  = (const int*)d_in[1];
  const int*   cams    = (const int*)d_in[2];
  const float* proxies = (const float*)d_in[3];
  const float* temp    = (const float*)d_in[4];

  const int B    = in_sizes[1];
  const int D    = in_sizes[0] / B;
  const int N    = in_sizes[3] / D;
  const int nseg = N * NUM_CAMS;

  // workspace carve-out
  size_t o = 0;
  auto take = [&](size_t bytes) { size_t p = o; o += (bytes + 15) & ~(size_t)15; return p; };
  char* base = (char*)d_ws;
  float*              own     = (float*)(base + take((size_t)B * 4));
  float*              sumexp  = (float*)(base + take((size_t)B * 4));
  unsigned long long* minpack = (unsigned long long*)(base + take((size_t)nseg * 8));
  float*              lossAcc = (float*)(base + take(32));
  size_t smallNeed = o;
  unsigned short* fhi = (unsigned short*)(base + take((size_t)B * D * 2));
  unsigned short* flo = (unsigned short*)(base + take((size_t)B * D * 2));
  unsigned short* phi = (unsigned short*)(base + take((size_t)N * D * 2));
  unsigned short* plo = (unsigned short*)(base + take((size_t)N * D * 2));
  const bool usePre = (ws_size >= o) && (smallNeed <= ws_size);

  int initN = nseg > B ? nseg : B;
  cam_proxy_init<<<(initN + 255) / 256, 256, 0, stream>>>(minpack, sumexp, lossAcc, nseg, B);

  if (usePre) {
    long f4 = (long)B * D / 4, p4 = (long)N * D / 4;
    cam_proxy_split<<<(unsigned)((f4 + 255) / 256), 256, 0, stream>>>(feats, fhi, flo, f4);
    cam_proxy_split<<<(unsigned)((p4 + 255) / 256), 256, 0, stream>>>(proxies, phi, plo, p4);
    dim3 grid((N + 255) / 256, (B + 127) / 128);
    cam_proxy_gemm_pre<<<grid, 256, 0, stream>>>(fhi, flo, phi, plo, temp, sumexp, B, N, D);
  } else {
    dim3 grid((N + 127) / 128, (B + 127) / 128);
    cam_proxy_gemm_fused<<<grid, 256, 0, stream>>>(feats, proxies, temp, sumexp, B, N, D);
  }

  cam_proxy_own<<<(B + 7) / 8, 256, 0, stream>>>(feats, proxies, labels, temp, own, B, D);
  cam_proxy_argmin<<<(B + 255) / 256, 256, 0, stream>>>(own, labels, cams, minpack, B);
  cam_proxy_pid<<<(N + 255) / 256, 256, 0, stream>>>(minpack, sumexp, own, temp, lossAcc, N);
  cam_proxy_final<<<1, 1, 0, stream>>>(lossAcc, (float*)d_out);
}
